// Model_75333726372102
// MI455X (gfx1250) — matrix operand order repaired and checked by reading, not compiled, not run on hardware
//
#include <hip/hip_runtime.h>
#include <cstdint>

// ---------------------------------------------------------------------------
// Model constants (from the reference)
// ---------------------------------------------------------------------------
#define V_  400
#define N_  8
#define CIN0 2
#define T_  5
#define K_  9
#define FS_ 14
#define TS2_ 48
#define KP1 10          // K+1
#define COLS 4000       // V * (K+1), contiguous axis of an A slice row
#define EPS_ 1e-5f

typedef __attribute__((ext_vector_type(16))) __bf16 v16bf;
typedef __attribute__((ext_vector_type(8)))  float  v8f;
typedef __attribute__((ext_vector_type(4)))  unsigned int u32x4;
typedef __attribute__((ext_vector_type(8)))  int i32x8;
typedef __attribute__((ext_vector_type(4)))  int i32x4;

// ---------------------------------------------------------------------------
// Semi-supervised branch (tiny): dense, propagate(A_mean@M)+leakyrelu, softmax
// ---------------------------------------------------------------------------
__global__ void lin_kernel(const float* __restrict__ in, const float* __restrict__ w,
                           const float* __restrict__ b, int rows, int kd, int ncol,
                           float* __restrict__ out) {
  int i = blockIdx.x * blockDim.x + threadIdx.x;
  if (i >= rows * ncol) return;
  int r = i / ncol, j = i - r * ncol;
  float acc = b[j];
  for (int k = 0; k < kd; ++k) acc += in[r * kd + k] * w[k * ncol + j];
  out[i] = acc;
}

__global__ void prop_kernel(const float* __restrict__ Amean, const float* __restrict__ Min,
                            int ncol, float* __restrict__ out) {
  int i = blockIdx.x * blockDim.x + threadIdx.x;
  if (i >= V_ * ncol) return;
  int v = i / ncol, j = i - v * ncol;
  float acc = 0.f;
  for (int u = 0; u < V_; ++u) acc += Amean[v * V_ + u] * Min[u * ncol + j];
  out[i] = acc > 0.f ? acc : 0.01f * acc;   // leaky_relu, slope 0.01
}

// softmax(Y2) -> zfac[v][10] = {1, Z[v][0..8]}; Yout = Y2[idx]
__global__ void post_kernel(const float* __restrict__ Y2, const int* __restrict__ idx,
                            float* __restrict__ zfac, float* __restrict__ Yout) {
  int v = blockIdx.x * blockDim.x + threadIdx.x;
  if (v >= V_) return;
  float y[K_]; float m = -1e30f;
  #pragma unroll
  for (int j = 0; j < K_; ++j) { y[j] = Y2[v * K_ + j]; m = fmaxf(m, y[j]); }
  float s = 0.f;
  #pragma unroll
  for (int j = 0; j < K_; ++j) { y[j] = __expf(y[j] - m); s += y[j]; }
  float inv = 1.f / s;
  zfac[v * KP1] = 1.f;
  #pragma unroll
  for (int j = 0; j < K_; ++j) zfac[v * KP1 + 1 + j] = y[j] * inv;
  int g = idx[v];
  #pragma unroll
  for (int j = 0; j < K_; ++j) Yout[v * K_ + j] = Y2[g * K_ + j];
}

// ---------------------------------------------------------------------------
// Per-layer stage 0: convert x (n,cin,t,V) -> zero-padded bf16 [nt][64][416]
// ---------------------------------------------------------------------------
__global__ void prep_xbf(const float* __restrict__ h, int cin, __bf16* __restrict__ xbf) {
  int nt = blockIdx.x;                    // 0..39
  int nn = nt / T_, tt = nt - nn * T_;
  for (int i = threadIdx.x; i < 64 * 416; i += blockDim.x) {
    int c = i / 416, u = i - c * 416;
    float v = 0.f;
    if (c < cin && u < V_) v = h[((size_t)(nn * cin + c) * T_ + tt) * V_ + u];
    xbf[(size_t)nt * 64 * 416 + i] = (__bf16)v;
  }
}

// ---------------------------------------------------------------------------
// Per-layer stage 1 (the hot loop):
//   H[c, col] = sum_u x[c,u] * A[s, u, col]      (col = v*10+k, contiguous)
//   G = relu(bn1(H * zfac[col]))  stored as (n, cin, t, 4000)
// WMMA bf16 16x16x32, K = u padded 400->416 (13 steps).
// A tiles (32 rows x 128 cols f32, LDS row stride 132) are staged by the
// Tensor Data Mover: wave 0 issues tensor_load_to_lds with a D# describing a
// 2D tile of the gathered A slice, padding 4 DWORDs per 128-DWORD row so the
// LDS layout matches as_[32][132].  Double-buffered; TENSORcnt + barrier sync.
// ---------------------------------------------------------------------------
template <int MT>
__global__ __launch_bounds__(256) void gcn_matmul_bn(
    const float* __restrict__ A, const int* __restrict__ x_t,
    const __bf16* __restrict__ xbf, const float* __restrict__ zfac,
    const float* __restrict__ g1p, const float* __restrict__ b1p,
    const float* __restrict__ m1p, const float* __restrict__ v1p,
    int cin, float* __restrict__ G) {
  __shared__ float as_[2][32][132];       // 32 u-rows x 128 cols, padded stride

  const int nt   = blockIdx.y;            // (n,t) pair
  const int s    = x_t[nt];               // gathered time slice
  const float* As = A + (size_t)s * ((size_t)V_ * COLS);
  const int colB = blockIdx.x * 128;      // block's 128-column window
  const int wave = threadIdx.x >> 5;
  const int lane = threadIdx.x & 31;
  const int lcol = lane & 15;
  const int kb   = (lane & 16) ? 8 : 0;   // A-frag K base for upper half-wave
  const int rsel = lane & 16;             // unused; B-frag rows are offset by kb
  const bool issuer = (threadIdx.x < 32); // wave 0 drives the TDM

  const __bf16* xr = xbf + (size_t)nt * 64 * 416;

  v8f acc[MT] = {};

  // One TDM descriptor per 32-row K-step. Rows clamped at the 400 boundary;
  // the final step's missing rows are zero-filled by the block before use.
  auto stage = [&](int step, int buf) {
#if defined(__HIP_DEVICE_COMPILE__)
    const float* gptr = As + (size_t)step * 32 * COLS + colB;
    unsigned long long ga = (unsigned long long)(size_t)gptr;
    unsigned ldsoff = (unsigned)(size_t)&as_[buf][0][0];
    int rows = V_ - step * 32; if (rows > 32) rows = 32;
    u32x4 g0;
    g0.x = 1u;                                           // count=1 (valid D#)
    g0.y = ldsoff;                                       // lds_addr (bytes)
    g0.z = (unsigned)ga;                                 // global_addr[31:0]
    g0.w = (unsigned)((ga >> 32) & 0x01ffffffu) | (2u << 30); // addr[56:32]|type=2
    i32x8 g1;
    g1[0] = 0x07920000;   // data_size=4B | pad_enable | pad_interval=128dw | pad_amount=4dw
    g1[1] = (int)((COLS & 0xffff) << 16);                // tensor_dim0 lo16
    g1[2] = (int)((COLS >> 16) | ((V_ & 0xffff) << 16)); // dim0 hi | tensor_dim1 lo16
    g1[3] = (int)((V_ >> 16) | (128 << 16));             // dim1 hi | tile_dim0=128
    g1[4] = rows;                                        // tile_dim1 (tile_dim2=0)
    g1[5] = COLS;                                        // tensor_dim0_stride lo32
    g1[6] = 0;                                           // stride hi | dim1_stride lo
    g1[7] = 0;
    i32x4 gz = {0, 0, 0, 0};
#if __clang_major__ >= 23
    i32x8 z8 = {};
    __builtin_amdgcn_tensor_load_to_lds(g0, g1, gz, gz, z8, 0);
#else
    __builtin_amdgcn_tensor_load_to_lds(g0, g1, gz, gz, 0);
#endif
#else
    (void)step; (void)buf;
#endif
  };

  if (issuer) {
    stage(0, 0);
#if defined(__HIP_DEVICE_COMPILE__)
    __builtin_amdgcn_s_wait_tensorcnt(0);
#endif
  }
  __syncthreads();

  for (int step = 0; step < 13; ++step) {       // 13*32 = 416 = 400 padded
    const int buf = step & 1;
    if (step + 1 < 13) {
      if (issuer) stage(step + 1, buf ^ 1);     // prefetch next tile via TDM
    } else {
      // last step: TDM delivered 16 valid rows; zero rows 16..31 of this buffer
      for (int i = threadIdx.x; i < 16 * 128; i += 256) {
        int r = 16 + (i >> 7), c = i & 127;
        as_[buf][r][c] = 0.f;
      }
      __syncthreads();
    }

    // B fragment: bf16 32x16, e -> K = kb + (e&7) + 2*(e&8), col = lane&15
    v16bf bfrag;
    #pragma unroll
    for (int e = 0; e < 16; ++e)
      bfrag[e] = (__bf16)as_[buf][kb + (e & 7) + ((e & 8) << 1)][wave * 16 + lcol];

    const int u0 = step * 32;
    #pragma unroll
    for (int mt = 0; mt < MT; ++mt) {
      // A fragment: bf16 16x32, M = lane&15, K per ISA half-wave split
      const __bf16* xp = xr + (size_t)(mt * 16 + lcol) * 416 + u0;
      v16bf afrag;
      #pragma unroll
      for (int e = 0; e < 8; ++e) afrag[e] = xp[kb + e];
      #pragma unroll
      for (int e = 0; e < 8; ++e) afrag[8 + e] = xp[16 + kb + e];
      acc[mt] = __builtin_amdgcn_wmma_f32_16x16x32_bf16(
          false, afrag, false, bfrag, (short)0, acc[mt], false, false);
    }

    if (issuer && step + 1 < 13) {
#if defined(__HIP_DEVICE_COMPILE__)
      __builtin_amdgcn_s_wait_tensorcnt(0);     // next tile landed in LDS
#endif
    }
    __syncthreads();
  }

  // Epilogue: zfac scale, bn1, relu, store G (n, cin, t, 4000)
  const int colN = colB + wave * 16 + lcol;
  const bool cv  = colN < COLS;
  const float zf = cv ? zfac[colN] : 0.f;
  const int nn = nt / T_, tt = nt - nn * T_;
  #pragma unroll
  for (int mt = 0; mt < MT; ++mt) {
    #pragma unroll
    for (int r = 0; r < 8; ++r) {
      int c = mt * 16 + r + ((lane & 16) >> 1);   // rows 0..7 / 8..15
      if (cv && c < cin) {
        float h = acc[mt][r] * zf;
        float g = (h - m1p[c]) * rsqrtf(v1p[c] + EPS_) * g1p[c] + b1p[c];
        G[((size_t)(nn * cin + c) * T_ + tt) * COLS + colN] = fmaxf(g, 0.f);
      }
    }
  }
}

// ---------------------------------------------------------------------------
// Per-layer stage 2: conv over (ci, t-1..t+1, k) + bias + bn2 + relu
//   out (n, cout, t, 400)
// ---------------------------------------------------------------------------
__global__ void gcn_conv_bn(const float* __restrict__ G,
                            const float* __restrict__ cw, const float* __restrict__ cb,
                            const float* __restrict__ g2, const float* __restrict__ b2,
                            const float* __restrict__ m2, const float* __restrict__ v2,
                            int cin, int cout, float* __restrict__ hout) {
  int nt = blockIdx.y;
  int nn = nt / T_, tt = nt - nn * T_;
  int vv = blockIdx.x * 32 + (threadIdx.x & 31);
  if (vv >= V_) return;
  for (int co = threadIdx.x >> 5; co < cout; co += 8) {
    float acc = cb[co];
    for (int ci = 0; ci < cin; ++ci) {
      #pragma unroll
      for (int dt = 0; dt < 3; ++dt) {
        int t2 = tt + dt - 1;
        if (t2 < 0 || t2 >= T_) continue;
        const float* gp = G + ((size_t)(nn * cin + ci) * T_ + t2) * COLS + vv * KP1;
        const float* wp = cw + ((size_t)(co * cin + ci) * 3 + dt) * KP1;
        #pragma unroll
        for (int k = 0; k < KP1; ++k) acc += gp[k] * wp[k];
      }
    }
    float bn = (acc - m2[co]) * rsqrtf(v2[co] + EPS_) * g2[co] + b2[co];
    hout[((size_t)(nn * cout + co) * T_ + tt) * V_ + vv] = fmaxf(bn, 0.f);
  }
}

// ---------------------------------------------------------------------------
// Final temporal conv (16->2 over all T) + residual last frame
// ---------------------------------------------------------------------------
__global__ void final_tconv(const float* __restrict__ h, const float* __restrict__ x,
                            const float* __restrict__ tw, const float* __restrict__ tb,
                            float* __restrict__ out) {
  int i = blockIdx.x * blockDim.x + threadIdx.x;
  if (i >= N_ * CIN0 * V_) return;
  int vv = i % V_; int rem = i / V_; int co = rem % CIN0; int nn = rem / CIN0;
  float acc = tb[co];
  for (int ci = 0; ci < 16; ++ci)
    #pragma unroll
    for (int dt = 0; dt < T_; ++dt)
      acc += h[((size_t)(nn * 16 + ci) * T_ + dt) * V_ + vv] * tw[(co * 16 + ci) * T_ + dt];
  acc += x[((size_t)(nn * CIN0 + co) * T_ + (T_ - 1)) * V_ + vv];  // last frame residual
  out[i] = acc;
}

// ---------------------------------------------------------------------------
// Host side
// ---------------------------------------------------------------------------
struct LayerP {
  const float *bn1g, *bn1b, *bn1m, *bn1v, *cw, *cb, *bn2g, *bn2b, *bn2m, *bn2v;
};

extern "C" void kernel_launch(void* const* d_in, const int* in_sizes, int n_in,
                              void* d_out, int out_size, void* d_ws, size_t ws_size,
                              hipStream_t stream) {
  (void)n_in; (void)out_size; (void)ws_size;
  const float *x, *A, *Amean, *F, *s1w, *s1b, *s2w, *s2b, *tw, *tb;
  const int *x_t, *idx;
  LayerP L[4];

  auto fp = [&](int i) { return (const float*)d_in[i]; };

  if (in_sizes[0] == V_) {
    // jax pytree sorted-key order: idx, params{A, A_mean, F, layers[...],
    //   semi1_b, semi1_w, semi2_b, semi2_w, tconv_b, tconv_w}, x, x_t
    idx = (const int*)d_in[0];
    A = fp(1); Amean = fp(2); F = fp(3);
    for (int l = 0; l < 4; ++l) {
      int b = 4 + 10 * l;  // sorted: bn1_b,bn1_g,bn1_m,bn1_v,bn2_b,bn2_g,bn2_m,bn2_v,conv_b,conv_w
      L[l].bn1b = fp(b + 0); L[l].bn1g = fp(b + 1); L[l].bn1m = fp(b + 2); L[l].bn1v = fp(b + 3);
      L[l].bn2b = fp(b + 4); L[l].bn2g = fp(b + 5); L[l].bn2m = fp(b + 6); L[l].bn2v = fp(b + 7);
      L[l].cb = fp(b + 8);  L[l].cw = fp(b + 9);
    }
    s1b = fp(44); s1w = fp(45); s2b = fp(46); s2w = fp(47); tb = fp(48); tw = fp(49);
    x = fp(50); x_t = (const int*)d_in[51];
  } else {
    // dict insertion order: x, x_t, idx, params{A, F, A_mean, semi1_w, semi1_b,
    //   semi2_w, semi2_b, tconv_w, tconv_b, layers[...]}
    x = fp(0); x_t = (const int*)d_in[1]; idx = (const int*)d_in[2];
    A = fp(3); F = fp(4); Amean = fp(5);
    s1w = fp(6); s1b = fp(7); s2w = fp(8); s2b = fp(9); tw = fp(10); tb = fp(11);
    for (int l = 0; l < 4; ++l) {
      int b = 12 + 10 * l;  // bn1_g,bn1_b,bn1_m,bn1_v,conv_w,conv_b,bn2_g,bn2_b,bn2_m,bn2_v
      L[l].bn1g = fp(b + 0); L[l].bn1b = fp(b + 1); L[l].bn1m = fp(b + 2); L[l].bn1v = fp(b + 3);
      L[l].cw = fp(b + 4);  L[l].cb = fp(b + 5);
      L[l].bn2g = fp(b + 6); L[l].bn2b = fp(b + 7); L[l].bn2m = fp(b + 8); L[l].bn2v = fp(b + 9);
    }
  }

  // Workspace layout (floats).  Total ~51.4 MB.
  float* ws = (float*)d_ws;
  float* zfac = ws;                         size_t off = 4096;
  float* tmp1 = ws + off;                   off += 6400;
  float* Y1   = ws + off;                   off += 6400;
  float* tmp2 = ws + off;                   off += 3600;
  float* Y2   = ws + off;                   off += 3600;
  __bf16* xbf = (__bf16*)(ws + off);        off += (size_t)40 * 64 * 416 / 2;  // bf16
  float* G    = ws + off;                   off += (size_t)N_ * 64 * T_ * COLS;
  float* hA   = ws + off;                   off += (size_t)N_ * 64 * T_ * V_;
  float* hB   = ws + off;

  float* out  = (float*)d_out;              // [0,6400) = out, [6400,10000) = Y

  // Semi-supervised branch
  lin_kernel <<<(V_ * 16 + 255) / 256, 256, 0, stream>>>(F, s1w, s1b, V_, FS_, 16, tmp1);
  prop_kernel<<<(V_ * 16 + 255) / 256, 256, 0, stream>>>(Amean, tmp1, 16, Y1);
  lin_kernel <<<(V_ * K_ + 255) / 256, 256, 0, stream>>>(Y1, s2w, s2b, V_, 16, K_, tmp2);
  prop_kernel<<<(V_ * K_ + 255) / 256, 256, 0, stream>>>(Amean, tmp2, K_, Y2);
  post_kernel<<<(V_ + 255) / 256, 256, 0, stream>>>(Y2, idx, zfac, out + N_ * CIN0 * V_);

  // 4 DGCN layers
  const int cins[4]  = {2, 32, 64, 32};
  const int couts[4] = {32, 64, 32, 16};
  const float* hin = x;
  float* hbufs[2] = {hA, hB};
  dim3 gA((COLS + 127) / 128, N_ * T_);
  dim3 gB((V_ + 31) / 32, N_ * T_);
  for (int l = 0; l < 4; ++l) {
    int cin = cins[l], cout = couts[l];
    prep_xbf<<<N_ * T_, 256, 0, stream>>>(hin, cin, xbf);
    switch ((cin + 15) / 16) {
      case 1: gcn_matmul_bn<1><<<gA, 256, 0, stream>>>(A, x_t, xbf, zfac,
                L[l].bn1g, L[l].bn1b, L[l].bn1m, L[l].bn1v, cin, G); break;
      case 2: gcn_matmul_bn<2><<<gA, 256, 0, stream>>>(A, x_t, xbf, zfac,
                L[l].bn1g, L[l].bn1b, L[l].bn1m, L[l].bn1v, cin, G); break;
      default: gcn_matmul_bn<4><<<gA, 256, 0, stream>>>(A, x_t, xbf, zfac,
                L[l].bn1g, L[l].bn1b, L[l].bn1m, L[l].bn1v, cin, G); break;
    }
    gcn_conv_bn<<<gB, 256, 0, stream>>>(G, L[l].cw, L[l].cb,
        L[l].bn2g, L[l].bn2b, L[l].bn2m, L[l].bn2v, cin, cout, hbufs[l & 1]);
    hin = hbufs[l & 1];
  }

  final_tconv<<<(N_ * CIN0 * V_ + 255) / 256, 256, 0, stream>>>(hin, x, tw, tb, out);
}
